// DecoderRNN_18013092839908
// MI455X (gfx1250) — matrix operand order repaired and checked by reading, not compiled, not run on hardware
//
#include <hip/hip_runtime.h>

typedef __attribute__((ext_vector_type(16))) __bf16 v16bf;
typedef __attribute__((ext_vector_type(8)))  float  v8f;
typedef unsigned int u32x4 __attribute__((ext_vector_type(4)));

#define BB   128
#define TT   128
#define SS   512
#define CC   34
#define KEYD 128
#define VALD 256
#define HIDD 512
#define H0D  768
#define BM   16
#define NTHREADS 256

// ---- explicit global-address-space accessors (avoid flat_load/flat_store) ----
typedef const __attribute__((address_space(1))) u32x4*  gc_u32x4p;
typedef const __attribute__((address_space(1))) float*  gc_floatp;
typedef       __attribute__((address_space(1))) float*  g_floatp;
typedef const __attribute__((address_space(1))) int*    gc_intp;

__device__ __forceinline__ float gloadf(const float* p) { return *(gc_floatp)(uintptr_t)p; }
__device__ __forceinline__ void  gstoref(float* p, float v) { *(g_floatp)(uintptr_t)p = v; }
__device__ __forceinline__ int   gloadi(const int* p) { return *(gc_intp)(uintptr_t)p; }

__device__ __forceinline__ float sigmoidf_(float x) { return 1.0f / (1.0f + __expf(-x)); }

__device__ __forceinline__ v8f vzero8() {
  v8f v;
  #pragma unroll
  for (int i = 0; i < 8; ++i) v[i] = 0.0f;
  return v;
}

// A fragment: 16x32 bf16 tile, row-major rows in LDS (address space inferred -> ds_load_b128).
// ISA 16-bit A layout: lane m=lane&15, half=lane>>4;
// VGPR0-3 cover k = half*8..+8, VGPR4-7 cover k = 16+half*8..+8
__device__ __forceinline__ v16bf load_a_frag(const __bf16* row, int kc, int lane) {
  const __bf16* p = row + kc * 32 + ((lane >> 4) << 3);
  v16bf f;
  u32x4* fp = reinterpret_cast<u32x4*>(&f);
  fp[0] = *reinterpret_cast<const u32x4*>(p);
  fp[1] = *reinterpret_cast<const u32x4*>(p + 16);
  return f;
}
// B fragment from GLOBAL weights: W row-major [N x K]; 'row' = this lane's N row.
// lane half h holds K 8h..8h+7 and 16+8h..16+8h+7 (matches A layout) -> global_load_b128 x2
__device__ __forceinline__ v16bf load_b_frag(const __bf16* row, int kc, int lane) {
  const __bf16* p = row + kc * 32 + ((lane >> 4) << 3);
  v16bf f;
  u32x4* fp = reinterpret_cast<u32x4*>(&f);
  fp[0] = *(gc_u32x4p)(uintptr_t)p;
  fp[1] = *(gc_u32x4p)(uintptr_t)(p + 16);
  return f;
}

__device__ __forceinline__ v8f wmma_bf16(v16bf a, v16bf b, v8f c) {
  return __builtin_amdgcn_wmma_f32_16x16x32_bf16(false, a, false, b, (short)0, c, false, false);
}

// workgroup-scope fence: producers/consumers share the WGP; no device-scope cache ops needed
__device__ __forceinline__ void wg_fence() {
  __builtin_amdgcn_fence(__ATOMIC_ACQ_REL, "workgroup");
}

template<int COLS, int STRIDE>
__device__ __forceinline__ void stage_bf16(__bf16* dst, const float* __restrict__ src, int tid) {
  for (int i = tid; i < 16 * COLS; i += NTHREADS) {
    int r = i / COLS;
    int c = i - r * COLS;
    dst[r * STRIDE + c] = (__bf16)gloadf(src + i);
  }
}

// ---- GRU layer with WMMA (generic: input DIN, hidden H). Updates st_h in place. ----
template<int H, int DIN, int AST, int HST>
__device__ void gru_layer(float* __restrict__ st_h,
                          const __bf16* __restrict__ wih,
                          const __bf16* __restrict__ whh,
                          const float* __restrict__ bih,
                          const float* __restrict__ bhh,
                          const __bf16* sAct, const __bf16* sHid,
                          int lane, int wave) {
  constexpr int NT = H / 16, KI = DIN / 32, KH = H / 32;
  const int n_ = lane & 15;
  const int mB = (lane >> 4) << 3;
  const __bf16* arow = sAct + n_ * AST;
  const __bf16* hrow = sHid + n_ * HST;
  for (int jt = wave; jt < NT; jt += 8) {
    const int j0 = jt * 16;
    const __bf16* wiR = wih + (size_t)(j0 + n_) * DIN;
    const __bf16* wiZ = wiR + (size_t)H * DIN;
    const __bf16* wiN = wiZ + (size_t)H * DIN;
    const __bf16* whR = whh + (size_t)(j0 + n_) * H;
    const __bf16* whZ = whR + (size_t)H * H;
    const __bf16* whN = whZ + (size_t)H * H;
    v8f iR = vzero8(), iZ = vzero8(), iN = vzero8();
    v8f hR = vzero8(), hZ = vzero8(), hN = vzero8();
    for (int kc = 0; kc < KI; ++kc) {
      v16bf af = load_a_frag(arow, kc, lane);
      iR = wmma_bf16(af, load_b_frag(wiR, kc, lane), iR);
      iZ = wmma_bf16(af, load_b_frag(wiZ, kc, lane), iZ);
      iN = wmma_bf16(af, load_b_frag(wiN, kc, lane), iN);
    }
    for (int kc = 0; kc < KH; ++kc) {
      v16bf hf = load_a_frag(hrow, kc, lane);
      hR = wmma_bf16(hf, load_b_frag(whR, kc, lane), hR);
      hZ = wmma_bf16(hf, load_b_frag(whZ, kc, lane), hZ);
      hN = wmma_bf16(hf, load_b_frag(whN, kc, lane), hN);
    }
    const int j = j0 + n_;
    const float bir = gloadf(bih + j), biz = gloadf(bih + H + j), bin_ = gloadf(bih + 2 * H + j);
    const float bhr = gloadf(bhh + j), bhz = gloadf(bhh + H + j), bhn = gloadf(bhh + 2 * H + j);
    #pragma unroll
    for (int r = 0; r < 8; ++r) {
      const int m = mB + r;
      float rg = sigmoidf_(iR[r] + bir + hR[r] + bhr);
      float zg = sigmoidf_(iZ[r] + biz + hZ[r] + bhz);
      float ng = tanhf(iN[r] + bin_ + rg * (hN[r] + bhn));
      float hp = gloadf(st_h + m * H + j);
      gstoref(st_h + m * H + j, (1.0f - zg) * ng + zg * hp);  // each (m,j) owned by one lane
    }
  }
}

// ---- Layer 0: gi is a one-hot gather from w_ih0; gh via WMMA (H=768). ----
__device__ void gru_layer0(float* __restrict__ st_h0,
                           const __bf16* __restrict__ whh0,
                           const float* __restrict__ w_ih0,
                           const float* __restrict__ bih, const float* __restrict__ bhh,
                           const __bf16* sHid, const int* sTok, int lane, int wave) {
  constexpr int H = 768, HST = 776;
  const int n_ = lane & 15, mB = (lane >> 4) << 3;
  const __bf16* hrow = sHid + n_ * HST;
  for (int jt = wave; jt < H / 16; jt += 8) {
    const int j0 = jt * 16;
    const __bf16* whR = whh0 + (size_t)(j0 + n_) * H;
    const __bf16* whZ = whR + (size_t)H * H;
    const __bf16* whN = whZ + (size_t)H * H;
    v8f hR = vzero8(), hZ = vzero8(), hN = vzero8();
    for (int kc = 0; kc < H / 32; ++kc) {
      v16bf hf = load_a_frag(hrow, kc, lane);
      hR = wmma_bf16(hf, load_b_frag(whR, kc, lane), hR);
      hZ = wmma_bf16(hf, load_b_frag(whZ, kc, lane), hZ);
      hN = wmma_bf16(hf, load_b_frag(whN, kc, lane), hN);
    }
    const int j = j0 + n_;
    const float bir = gloadf(bih + j), biz = gloadf(bih + H + j), bin_ = gloadf(bih + 2 * H + j);
    const float bhr = gloadf(bhh + j), bhz = gloadf(bhh + H + j), bhn = gloadf(bhh + 2 * H + j);
    #pragma unroll
    for (int r = 0; r < 8; ++r) {
      const int m = mB + r;
      const int tok = sTok[m];
      float gir = gloadf(w_ih0 + j * CC + tok) + bir;            // one-hot x -> column gather
      float giz = gloadf(w_ih0 + (H + j) * CC + tok) + biz;
      float gin = gloadf(w_ih0 + (2 * H + j) * CC + tok) + bin_;
      float rg = sigmoidf_(gir + hR[r] + bhr);
      float zg = sigmoidf_(giz + hZ[r] + bhz);
      float ng = tanhf(gin + rg * (hN[r] + bhn));
      float hp = gloadf(st_h0 + m * H + j);
      gstoref(st_h0 + m * H + j, (1.0f - zg) * ng + zg * hp);
    }
  }
}

__global__ void f32_to_bf16(const float* __restrict__ src, __bf16* __restrict__ dst, int n) {
  int stride = gridDim.x * blockDim.x;
  for (int i = blockIdx.x * blockDim.x + threadIdx.x; i < n; i += stride)
    dst[i] = (__bf16)gloadf(src + i);
}

__global__ __launch_bounds__(NTHREADS, 1)
void decoder_rnn(const int* __restrict__ seq, const int* __restrict__ masks,
                 const float* __restrict__ keys, const float* __restrict__ values,
                 const float* __restrict__ first_hidden,
                 const float* __restrict__ init_h1, const float* __restrict__ init_h2,
                 const float* __restrict__ init_h3,
                 const float* __restrict__ qw, const float* __restrict__ qb,
                 const float* __restrict__ sw, const float* __restrict__ sb,
                 const float* __restrict__ w_ih0,
                 const float* __restrict__ b_ih0, const float* __restrict__ b_hh0,
                 const float* __restrict__ b_ih1, const float* __restrict__ b_hh1,
                 const float* __restrict__ b_ih2, const float* __restrict__ b_hh2,
                 const float* __restrict__ b_ih3, const float* __restrict__ b_hh3,
                 const __bf16* __restrict__ whh0, const __bf16* __restrict__ wih1,
                 const __bf16* __restrict__ whh1, const __bf16* __restrict__ wih2,
                 const __bf16* __restrict__ whh2, const __bf16* __restrict__ wih3,
                 const __bf16* __restrict__ whh3,
                 float* __restrict__ states, float* __restrict__ out) {
  __shared__ __align__(16) union {
    struct { __bf16 act[16 * 776]; __bf16 hid[16 * 776]; } mm;   // ~48.5 KB
    struct { float att[16 * SS]; float q[KEYD]; } at;            // ~32.5 KB
  } sh;
  __shared__ int sTok[16];

  const int tid = threadIdx.x;
  const int lane = tid & 31, wave = tid >> 5;
  const int rows = blockIdx.x * BM;

  float* st_h0 = states + (size_t)blockIdx.x * (16 * H0D + 3 * 16 * HIDD);
  float* st_h1 = st_h0 + 16 * H0D;
  float* st_h2 = st_h1 + 16 * HIDD;
  float* st_h3 = st_h2 + 16 * HIDD;

  // ---------- attention (t = 0 only; t>0 attend() is dead code) ----------
  for (int k = tid; k < KEYD; k += NTHREADS) {
    float a = gloadf(qb + k);
    const float* qr = qw + k * HIDD;
    for (int h = 0; h < HIDD; ++h) a += gloadf(init_h3 + h) * gloadf(qr + h);
    sh.at.q[k] = a;
  }
  __syncthreads();
  for (int p = tid; p < 16 * SS; p += NTHREADS) {
    int b = p >> 9, s = p & (SS - 1);
    const float* kp = keys + ((size_t)s * BB + rows + b) * KEYD;
    float e = 0.f;
    for (int k = 0; k < KEYD; ++k) e += sh.at.q[k] * gloadf(kp + k);
    sh.at.att[p] = (s < gloadi(masks + rows + b)) ? e : 0.f;  // multiplicative mask pre-softmax
  }
  __syncthreads();
  if (tid < 16) {
    float* e = sh.at.att + tid * SS;
    float mx = -3.4e38f;
    for (int s = 0; s < SS; ++s) mx = fmaxf(mx, e[s]);
    float sum = 0.f;
    for (int s = 0; s < SS; ++s) { float v = __expf(e[s] - mx); e[s] = v; sum += v; }
    float inv = 1.f / sum;
    for (int s = 0; s < SS; ++s) e[s] *= inv;
  }
  __syncthreads();
  for (int p = tid; p < 16 * VALD; p += NTHREADS) {
    int b = p >> 8, v = p & (VALD - 1);
    const float* ar = sh.at.att + b * SS;
    float a = 0.f;
    for (int s = 0; s < SS; ++s)
      a += ar[s] * gloadf(values + ((size_t)s * BB + rows + b) * VALD + v);
    gstoref(st_h0 + b * H0D + v, a);                       // ctx -> h0[:, 0:256]
  }
  for (int p = tid; p < 16 * HIDD; p += NTHREADS) {
    int b = p >> 9, h = p & (HIDD - 1);
    gstoref(st_h0 + b * H0D + VALD + h, gloadf(first_hidden + h));  // h0[:, 256:768]
    gstoref(st_h1 + b * HIDD + h, gloadf(init_h1 + h));
    gstoref(st_h2 + b * HIDD + h, gloadf(init_h2 + h));
    gstoref(st_h3 + b * HIDD + h, gloadf(init_h3 + h));
  }
  wg_fence();
  __syncthreads();

  // ---------- serial time loop, all 4 layers fused per block ----------
  for (int t = 0; t < TT; ++t) {
    if (tid < 16) sTok[tid] = gloadi(seq + (rows + tid) * TT + t);
    stage_bf16<H0D, 776>(sh.mm.hid, st_h0, tid);
    __syncthreads();
    gru_layer0(st_h0, whh0, w_ih0, b_ih0, b_hh0, sh.mm.hid, sTok, lane, wave);
    wg_fence(); __syncthreads();

    stage_bf16<H0D, 776>(sh.mm.act, st_h0, tid);
    stage_bf16<HIDD, 520>(sh.mm.hid, st_h1, tid);
    __syncthreads();
    gru_layer<HIDD, H0D, 776, 520>(st_h1, wih1, whh1, b_ih1, b_hh1, sh.mm.act, sh.mm.hid, lane, wave);
    wg_fence(); __syncthreads();

    stage_bf16<HIDD, 520>(sh.mm.act, st_h1, tid);
    stage_bf16<HIDD, 520>(sh.mm.hid, st_h2, tid);
    __syncthreads();
    gru_layer<HIDD, HIDD, 520, 520>(st_h2, wih2, whh2, b_ih2, b_hh2, sh.mm.act, sh.mm.hid, lane, wave);
    wg_fence(); __syncthreads();

    stage_bf16<HIDD, 520>(sh.mm.act, st_h2, tid);
    stage_bf16<HIDD, 520>(sh.mm.hid, st_h3, tid);
    __syncthreads();
    gru_layer<HIDD, HIDD, 520, 520>(st_h3, wih3, whh3, b_ih3, b_hh3, sh.mm.act, sh.mm.hid, lane, wave);
    wg_fence(); __syncthreads();

    // scores[t] = sigmoid(h3) @ sw.T + sb   (16x34 dots over 512)
    for (int p = tid; p < 16 * CC; p += NTHREADS) {
      int b = p / CC, c = p - (p / CC) * CC;
      const float* hr = st_h3 + b * HIDD;
      const float* swr = sw + c * HIDD;
      float a = gloadf(sb + c);
      for (int h = 0; h < HIDD; ++h) a += sigmoidf_(gloadf(hr + h)) * gloadf(swr + h);
      gstoref(out + ((size_t)(rows + b) * TT + t) * CC + c, a);
    }
    __syncthreads();
  }
}

extern "C" void kernel_launch(void* const* d_in, const int* in_sizes, int n_in,
                              void* d_out, int out_size, void* d_ws, size_t ws_size,
                              hipStream_t stream) {
  const int*   seq    = (const int*)d_in[0];
  const int*   masks  = (const int*)d_in[1];
  const float* keys   = (const float*)d_in[2];
  const float* values = (const float*)d_in[3];
  const float* first_hidden = (const float*)d_in[4];
  // d_in[5] (init_h0) is unused by the reference forward pass
  const float* init_h1 = (const float*)d_in[6];
  const float* init_h2 = (const float*)d_in[7];
  const float* init_h3 = (const float*)d_in[8];
  const float* qw = (const float*)d_in[9];
  const float* qb = (const float*)d_in[10];
  const float* sw = (const float*)d_in[11];
  const float* sb = (const float*)d_in[12];
  const float* w_ih0 = (const float*)d_in[13];
  const float* w_hh0 = (const float*)d_in[14];
  const float* b_ih0 = (const float*)d_in[15];
  const float* b_hh0 = (const float*)d_in[16];
  const float* w_ih1 = (const float*)d_in[17];
  const float* w_hh1 = (const float*)d_in[18];
  const float* b_ih1 = (const float*)d_in[19];
  const float* b_hh1 = (const float*)d_in[20];
  const float* w_ih2 = (const float*)d_in[21];
  const float* w_hh2 = (const float*)d_in[22];
  const float* b_ih2 = (const float*)d_in[23];
  const float* b_hh2 = (const float*)d_in[24];
  const float* w_ih3 = (const float*)d_in[25];
  const float* w_hh3 = (const float*)d_in[26];
  const float* b_ih3 = (const float*)d_in[27];
  const float* b_hh3 = (const float*)d_in[28];

  // bf16 weight workspace layout (13.8 MB) + fp32 state (1.2 MB)
  __bf16* bf = (__bf16*)d_ws;
  const size_t n_whh0 = (size_t)2304 * 768;
  const size_t n_wih1 = (size_t)1536 * 768;
  const size_t n_whh  = (size_t)1536 * 512;
  __bf16* whh0b = bf;
  __bf16* wih1b = whh0b + n_whh0;
  __bf16* whh1b = wih1b + n_wih1;
  __bf16* wih2b = whh1b + n_whh;
  __bf16* whh2b = wih2b + n_whh;
  __bf16* wih3b = whh2b + n_whh;
  __bf16* whh3b = wih3b + n_whh;
  size_t bfEnd = (size_t)((whh3b + n_whh) - bf) * sizeof(__bf16);
  float* states = (float*)((char*)d_ws + ((bfEnd + 255) & ~(size_t)255));

  auto conv = [&](const float* s, __bf16* d, size_t n) {
    int blocks = (int)((n + 255) / 256); if (blocks > 2048) blocks = 2048;
    f32_to_bf16<<<blocks, 256, 0, stream>>>(s, d, (int)n);
  };
  conv(w_hh0, whh0b, n_whh0);
  conv(w_ih1, wih1b, n_wih1);
  conv(w_hh1, whh1b, n_whh);
  conv(w_ih2, wih2b, n_whh);
  conv(w_hh2, whh2b, n_whh);
  conv(w_ih3, wih3b, n_whh);
  conv(w_hh3, whh3b, n_whh);

  decoder_rnn<<<BB / BM, NTHREADS, 0, stream>>>(
      seq, masks, keys, values, first_hidden, init_h1, init_h2, init_h3,
      qw, qb, sw, sb, w_ih0,
      b_ih0, b_hh0, b_ih1, b_hh1, b_ih2, b_hh2, b_ih3, b_hh3,
      whh0b, wih1b, whh1b, wih2b, whh2b, wih3b, whh3b,
      states, (float*)d_out);

  (void)in_sizes; (void)n_in; (void)out_size; (void)ws_size;
}
